// ContMultiHeadedAttention_63531156243024
// MI455X (gfx1250) — matrix operand order repaired and checked by reading, not compiled, not run on hardware
//
#include <hip/hip_runtime.h>

typedef __attribute__((ext_vector_type(16))) _Float16 v16h;
typedef __attribute__((ext_vector_type(8)))  _Float16 h8;
typedef __attribute__((ext_vector_type(8)))  float    v8f;

__device__ __forceinline__ v16h cat8(h8 lo, h8 hi) {
    return __builtin_shufflevector(lo, hi, 0,1,2,3,4,5,6,7,8,9,10,11,12,13,14,15);
}
__device__ __forceinline__ v8f wmma_f16(v16h a, v16h b, v8f c) {
    return __builtin_amdgcn_wmma_f32_16x16x32_f16(false, a, false, b, (short)0, c, false, false);
}
// Async global->LDS copy, 16 bytes per lane (ASYNCcnt-tracked).
// GVS addressing: mem = saddr64 + voff32, LDS dest = ldsoff32.
__device__ __forceinline__ void async_copy_b128(unsigned ldsoff, unsigned voff, const void* base) {
    asm volatile("global_load_async_to_lds_b128 %0, %1, %2"
                 :: "v"(ldsoff), "v"(voff), "s"(base) : "memory");
}
__device__ __forceinline__ void wait_async0() {
    asm volatile("s_wait_asynccnt 0x0" ::: "memory");
}

// ---------------------------------------------------------------------------
// Projection GEMM: Y = X.W^T + b ; X [4096,1024] f32, W [1024,1024] f32.
// f16 output in head-split layout: Y[((b*16 + n>>6)*2048 + l)*outStride + outOffset + (n&63)]
// 128x128x32 tiles, 8 waves, 2x4 16x16 tiles per wave; register double-buffer.
// ---------------------------------------------------------------------------
__global__ __launch_bounds__(256)
void proj_kernel(const float* __restrict__ X, const float* __restrict__ W,
                 const float* __restrict__ bias, _Float16* __restrict__ Y,
                 int outStride, int outOffset)
{
    constexpr int K = 1024;
    __shared__ __align__(16) _Float16 As[128][40];
    __shared__ __align__(16) _Float16 Bs[128][40];

    const int tid  = threadIdx.x;
    const int lane = tid & 31;
    const int wv   = tid >> 5;
    const int m0   = blockIdx.y * 128;
    const int n0   = blockIdx.x * 128;
    const int wr   = (wv & 3) * 32;
    const int wc   = (wv >> 2) * 64;
    const int half = lane >> 4;
    const int colL = lane & 15;

    v8f acc[2][4];
    #pragma unroll
    for (int i = 0; i < 2; ++i)
        #pragma unroll
        for (int j = 0; j < 4; ++j)
            #pragma unroll
            for (int r = 0; r < 8; ++r) acc[i][j][r] = 0.0f;

    const int ldRow = tid >> 1;
    const int ldCol = (tid & 1) * 16;
    const float4* xa = (const float4*)(X + (size_t)(m0 + ldRow) * K + ldCol);
    const float4* wa = (const float4*)(W + (size_t)(n0 + ldRow) * K + ldCol);

    float4 ra[4], rb[4];
    #pragma unroll
    for (int i = 0; i < 4; ++i) { ra[i] = xa[i]; rb[i] = wa[i]; }

    for (int k0 = 0; k0 < K; k0 += 32) {
        #pragma unroll
        for (int i = 0; i < 4; ++i) {
            As[ldRow][ldCol + 4*i + 0] = (_Float16)ra[i].x;
            As[ldRow][ldCol + 4*i + 1] = (_Float16)ra[i].y;
            As[ldRow][ldCol + 4*i + 2] = (_Float16)ra[i].z;
            As[ldRow][ldCol + 4*i + 3] = (_Float16)ra[i].w;
            Bs[ldRow][ldCol + 4*i + 0] = (_Float16)rb[i].x;
            Bs[ldRow][ldCol + 4*i + 1] = (_Float16)rb[i].y;
            Bs[ldRow][ldCol + 4*i + 2] = (_Float16)rb[i].z;
            Bs[ldRow][ldCol + 4*i + 3] = (_Float16)rb[i].w;
        }
        __syncthreads();

        const bool more = (k0 + 32) < K;
        float4 na[4], nb[4];
        if (more) {
            const float4* xn = (const float4*)((const float*)xa + (k0 + 32));
            const float4* wn = (const float4*)((const float*)wa + (k0 + 32));
            #pragma unroll
            for (int i = 0; i < 4; ++i) { na[i] = xn[i]; nb[i] = wn[i]; }
        }

        const int klo   = half * 8;
        const int koffB = half * 8;
        const int ar0   = wr + colL;
        v16h a0 = cat8(*(const h8*)&As[ar0     ][klo], *(const h8*)&As[ar0     ][klo + 16]);
        v16h a1 = cat8(*(const h8*)&As[ar0 + 16][klo], *(const h8*)&As[ar0 + 16][klo + 16]);
        #pragma unroll
        for (int t = 0; t < 4; ++t) {
            const int bc = wc + t*16 + colL;
            v16h bt = cat8(*(const h8*)&Bs[bc][koffB], *(const h8*)&Bs[bc][koffB + 16]);
            acc[0][t] = wmma_f16(a0, bt, acc[0][t]);
            acc[1][t] = wmma_f16(a1, bt, acc[1][t]);
        }
        __syncthreads();
        if (more) {
            #pragma unroll
            for (int i = 0; i < 4; ++i) { ra[i] = na[i]; rb[i] = nb[i]; }
        }
    }

    #pragma unroll
    for (int tm = 0; tm < 2; ++tm)
        #pragma unroll
        for (int t = 0; t < 4; ++t) {
            const int n = n0 + wc + t*16 + colL;
            const int h = n >> 6, d = n & 63;
            const float bv = bias[n];
            #pragma unroll
            for (int r = 0; r < 8; ++r) {
                const int m = m0 + wr + tm*16 + r + half*8;
                const int b = m >> 11, l = m & 2047;      // L = 2048
                Y[((size_t)(b*16 + h) * 2048 + l) * outStride + outOffset + d] =
                    (_Float16)(acc[tm][t][r] + bv);
            }
        }
}

// ---------------------------------------------------------------------------
// Output projection: out = ctx_f16 . Wo^T + bo  (f32, row-major [4096,1024])
// ---------------------------------------------------------------------------
__global__ __launch_bounds__(256)
void oproj_kernel(const _Float16* __restrict__ A, const float* __restrict__ W,
                  const float* __restrict__ bias, float* __restrict__ Y)
{
    constexpr int K = 1024, N = 1024;
    __shared__ __align__(16) _Float16 As[128][40];
    __shared__ __align__(16) _Float16 Bs[128][40];

    const int tid  = threadIdx.x;
    const int lane = tid & 31;
    const int wv   = tid >> 5;
    const int m0   = blockIdx.y * 128;
    const int n0   = blockIdx.x * 128;
    const int wr   = (wv & 3) * 32;
    const int wc   = (wv >> 2) * 64;
    const int half = lane >> 4;
    const int colL = lane & 15;

    v8f acc[2][4];
    #pragma unroll
    for (int i = 0; i < 2; ++i)
        #pragma unroll
        for (int j = 0; j < 4; ++j)
            #pragma unroll
            for (int r = 0; r < 8; ++r) acc[i][j][r] = 0.0f;

    const int ldRow = tid >> 1;
    const int ldCol = (tid & 1) * 16;
    const _Float16* aa = A + (size_t)(m0 + ldRow) * K + ldCol;
    const float4*   wa = (const float4*)(W + (size_t)(n0 + ldRow) * K + ldCol);

    h8 ra0 = *(const h8*)&aa[0], ra1 = *(const h8*)&aa[8];
    float4 rb[4];
    #pragma unroll
    for (int i = 0; i < 4; ++i) rb[i] = wa[i];

    for (int k0 = 0; k0 < K; k0 += 32) {
        *(h8*)&As[ldRow][ldCol]     = ra0;
        *(h8*)&As[ldRow][ldCol + 8] = ra1;
        #pragma unroll
        for (int i = 0; i < 4; ++i) {
            Bs[ldRow][ldCol + 4*i + 0] = (_Float16)rb[i].x;
            Bs[ldRow][ldCol + 4*i + 1] = (_Float16)rb[i].y;
            Bs[ldRow][ldCol + 4*i + 2] = (_Float16)rb[i].z;
            Bs[ldRow][ldCol + 4*i + 3] = (_Float16)rb[i].w;
        }
        __syncthreads();

        const bool more = (k0 + 32) < K;
        h8 nextA0, nextA1; float4 nb[4];
        if (more) {
            const _Float16* an = aa + (k0 + 32);
            nextA0 = *(const h8*)&an[0];
            nextA1 = *(const h8*)&an[8];
            const float4* wn = (const float4*)((const float*)wa + (k0 + 32));
            #pragma unroll
            for (int i = 0; i < 4; ++i) nb[i] = wn[i];
        }

        const int klo   = half * 8;
        const int koffB = half * 8;
        const int ar0   = wr + colL;
        v16h a0 = cat8(*(const h8*)&As[ar0     ][klo], *(const h8*)&As[ar0     ][klo + 16]);
        v16h a1 = cat8(*(const h8*)&As[ar0 + 16][klo], *(const h8*)&As[ar0 + 16][klo + 16]);
        #pragma unroll
        for (int t = 0; t < 4; ++t) {
            const int bc = wc + t*16 + colL;
            v16h bt = cat8(*(const h8*)&Bs[bc][koffB], *(const h8*)&Bs[bc][koffB + 16]);
            acc[0][t] = wmma_f16(a0, bt, acc[0][t]);
            acc[1][t] = wmma_f16(a1, bt, acc[1][t]);
        }
        __syncthreads();
        if (more) {
            ra0 = nextA0; ra1 = nextA1;
            #pragma unroll
            for (int i = 0; i < 4; ++i) rb[i] = nb[i];
        }
    }

    #pragma unroll
    for (int tm = 0; tm < 2; ++tm)
        #pragma unroll
        for (int t = 0; t < 4; ++t) {
            const int n = n0 + wc + t*16 + colL;
            const float bv = bias[n];
            #pragma unroll
            for (int r = 0; r < 8; ++r) {
                const int m = m0 + wr + tm*16 + r + half*8;
                Y[(size_t)m * N + n] = acc[tm][t][r] + bv;
            }
        }
}

// k_cat[row, 0:64] += k_cat[row, 64:128]   -> [kc+kp | kc]
__global__ void combine_kcat(_Float16* kcat, int rows)
{
    int i = blockIdx.x * 256 + threadIdx.x;
    if (i >= rows * 64) return;
    int r = i >> 6, d = i & 63;
    size_t base = (size_t)r * 128;
    kcat[base + d] = (_Float16)((float)kcat[base + d] + (float)kcat[base + 64 + d]);
}

// gate[row] = sigmoid( q_c[row,:] . Wg + bg )
__global__ void gate_kernel(const _Float16* __restrict__ qcat,
                            const float* __restrict__ Wg, const float* __restrict__ bg,
                            float* __restrict__ gate, int rows)
{
    int i = blockIdx.x * 256 + threadIdx.x;
    if (i >= rows) return;
    const _Float16* row = qcat + (size_t)i * 128;
    float s = 0.0f;
    #pragma unroll
    for (int d = 0; d < 64; ++d) s += (float)row[d] * Wg[d];
    s += bg[0];
    gate[i] = 1.0f / (1.0f + __expf(-s));
}

// ---------------------------------------------------------------------------
// Flash attention. Q frags in registers; K tiles double-buffered in LDS via
// async global->LDS b128 (ASYNCcnt); VT + P single-buffered.
// LDS: 2x K(64x136) + VT(64x72) + P(128x72) = 62,464 B.
// ---------------------------------------------------------------------------
__global__ __launch_bounds__(256)
void attn_kernel(const _Float16* __restrict__ Qcat, const _Float16* __restrict__ Kcat,
                 const _Float16* __restrict__ Vh, const float* __restrict__ gate,
                 const float* __restrict__ sbias, const unsigned char* __restrict__ maskp,
                 _Float16* __restrict__ ctx)
{
    constexpr int Hn = 16, Ln = 2048;
    constexpr float scale = 0.125f;
    constexpr int KSTR = 136, PSTR = 72;
    constexpr unsigned KBYTES = 64 * KSTR * 2;                // 17408
    constexpr unsigned VTBYTES = 64 * 72 * 2;                 // 9216
    __shared__ __align__(16) char smem[2*KBYTES + VTBYTES + 128*PSTR*2];
    _Float16* kb0 = (_Float16*)smem;
    _Float16* kb1 = (_Float16*)(smem + KBYTES);
    _Float16* VT  = (_Float16*)(smem + 2*KBYTES);
    _Float16* Pb  = (_Float16*)(smem + 2*KBYTES + VTBYTES);

    const int tid  = threadIdx.x;
    const int lane = tid & 31;
    const int wv   = tid >> 5;
    const int q0   = blockIdx.x * 128;
    const int h    = blockIdx.y;
    const int b    = blockIdx.z;
    const int half = lane >> 4;
    const int colL = lane & 15;
    const int qrb  = wv * 16;

    const size_t headQK = (size_t)(b * Hn + h) * Ln * 128;
    const size_t headV  = (size_t)(b * Hn + h) * Ln * 64;
    const _Float16* kbase = Kcat + headQK;

    // Q fragments: 4 k-steps of 32 over d=128, held in registers
    v16h qf[4];
    {
        const _Float16* qsrc = Qcat + headQK + (size_t)(q0 + qrb + colL) * 128;
        #pragma unroll
        for (int ds = 0; ds < 4; ++ds) {
            const int klo = ds*32 + half*8;
            qf[ds] = cat8(*(const h8*)&qsrc[klo], *(const h8*)&qsrc[klo + 16]);
        }
    }

    // per-thread K staging geometry
    const int krow = tid >> 2;          // 0..63 keys
    const int kcb  = (tid & 3) * 32;    // 32 halves = 4 x b128

    // preload K block 0 (async into kb0)
    {
        const unsigned voff = (unsigned)((krow * 128 + kcb) * 2);
        const unsigned loff = (unsigned)((krow * KSTR + kcb) * 2);
        #pragma unroll
        for (int c = 0; c < 4; ++c)
            async_copy_b128(loff + c*16, voff + c*16, kbase);
    }
    wait_async0();
    __syncthreads();

    v8f o[4];
    float mrow[8], lrow[8];
    #pragma unroll
    for (int t = 0; t < 4; ++t)
        #pragma unroll
        for (int r = 0; r < 8; ++r) o[t][r] = 0.0f;
    #pragma unroll
    for (int r = 0; r < 8; ++r) { mrow[r] = -3.0e38f; lrow[r] = 0.0f; }

    int cur = 0;
    for (int kb = 0; kb < Ln; kb += 64) {
        _Float16* kc = cur ? kb1 : kb0;

        // (A) async prefetch K[kb+64] into the other buffer
        if (kb + 64 < Ln) {
            const unsigned voff = (unsigned)(((kb + 64 + krow) * 128 + kcb) * 2);
            const unsigned loff = (unsigned)((cur ^ 1) * KBYTES + (krow * KSTR + kcb) * 2);
            #pragma unroll
            for (int c = 0; c < 4; ++c)
                async_copy_b128(loff + c*16, voff + c*16, kbase);
        }
        // prefetch next spatial_bias block into cache
        __builtin_prefetch(&sbias[(size_t)b*Ln*Ln +
                                  (size_t)(q0 + qrb + half*8)*Ln + (kb + 64 + colL) % Ln], 0, 1);

        // (B) stage V[kb] transposed: VT[d][key]
        {
            const int key = tid >> 2;
            const int db  = (tid & 3) * 16;
            const _Float16* src = Vh + headV + (size_t)(kb + key) * 64 + db;
            h8 v0 = *(const h8*)&src[0];
            h8 v1 = *(const h8*)&src[8];
            #pragma unroll
            for (int i = 0; i < 8; ++i) {
                VT[(db + i) * 72 + key]     = v0[i];
                VT[(db + 8 + i) * 72 + key] = v1[i];
            }
        }

        // (C) S = Q.K^T over d=128
        v8f s[4];
        #pragma unroll
        for (int t = 0; t < 4; ++t)
            #pragma unroll
            for (int r = 0; r < 8; ++r) s[t][r] = 0.0f;
        #pragma unroll
        for (int ds = 0; ds < 4; ++ds) {
            const int koff = ds*32 + half*8;
            #pragma unroll
            for (int t = 0; t < 4; ++t) {
                const int key = t*16 + colL;
                v16h bt = cat8(*(const h8*)&kc[key * KSTR + koff],
                               *(const h8*)&kc[key * KSTR + koff + 16]);
                s[t] = wmma_f16(qf[ds], bt, s[t]);
            }
        }

        // (D) epilogue: scale + gate*spatial_bias + mask
        float g[8];
        #pragma unroll
        for (int r = 0; r < 8; ++r)
            g[r] = gate[(size_t)(b * Hn + h) * Ln + q0 + qrb + r + half*8];
        #pragma unroll
        for (int t = 0; t < 4; ++t) {
            const int kcol = kb + t*16 + colL;
            #pragma unroll
            for (int r = 0; r < 8; ++r) {
                const int q = q0 + qrb + r + half*8;
                const size_t bidx = (size_t)b * Ln * Ln + (size_t)q * Ln + kcol;
                float sv = s[t][r] * scale + g[r] * sbias[bidx];
                if (!maskp[bidx]) sv = -3.0e38f;
                s[t][r] = sv;
            }
        }

        // (E) online softmax; P into Pb (wave-private rows); rescale O
        float alpha[8];
        #pragma unroll
        for (int r = 0; r < 8; ++r) {
            float mx = fmaxf(fmaxf(s[0][r], s[1][r]), fmaxf(s[2][r], s[3][r]));
            #pragma unroll
            for (int off = 8; off >= 1; off >>= 1)
                mx = fmaxf(mx, __shfl_xor(mx, off, 32));
            const float mn = fmaxf(mrow[r], mx);
            alpha[r] = __expf(mrow[r] - mn);
            float sum = 0.0f;
            #pragma unroll
            for (int t = 0; t < 4; ++t) {
                const float p = __expf(s[t][r] - mn);
                s[t][r] = p;
                sum += p;
            }
            #pragma unroll
            for (int off = 8; off >= 1; off >>= 1)
                sum += __shfl_xor(sum, off, 32);
            lrow[r] = lrow[r] * alpha[r] + sum;
            mrow[r] = mn;
        }
        #pragma unroll
        for (int t = 0; t < 4; ++t)
            #pragma unroll
            for (int r = 0; r < 8; ++r) {
                Pb[(qrb + r + half*8) * PSTR + t*16 + colL] = (_Float16)s[t][r];
                o[t][r] *= alpha[r];
            }

        // (F) publish VT (and P) to all waves
        __syncthreads();

        // (G) O += P.V  (keys as reduction dim, 2 k-steps of 32)
        #pragma unroll
        for (int ks = 0; ks < 2; ++ks) {
            const int klo  = ks*32 + half*8;
            const int arow = qrb + colL;
            v16h a = cat8(*(const h8*)&Pb[arow * PSTR + klo],
                          *(const h8*)&Pb[arow * PSTR + klo + 16]);
            const int koff = ks*32 + half*8;
            #pragma unroll
            for (int t = 0; t < 4; ++t) {
                const int dcol = t*16 + colL;
                v16h bt = cat8(*(const h8*)&VT[dcol * 72 + koff],
                               *(const h8*)&VT[dcol * 72 + koff + 16]);
                o[t] = wmma_f16(a, bt, o[t]);
            }
        }

        // (H) K[kb+64] landed; VT free for next iteration
        wait_async0();
        __syncthreads();
        cur ^= 1;
    }

    // finalize: divide by l, store ctx f16 in [B, L, H*64]
    #pragma unroll
    for (int t = 0; t < 4; ++t)
        #pragma unroll
        for (int r = 0; r < 8; ++r) {
            const int q = q0 + qrb + r + half*8;
            const int d = t*16 + colL;
            const float val = o[t][r] / fmaxf(lrow[r], 1e-20f);
            ctx[((size_t)(b * Ln + q) * Hn + h) * 64 + d] = (_Float16)val;
        }
}

// ---------------------------------------------------------------------------
extern "C" void kernel_launch(void* const* d_in, const int* in_sizes, int n_in,
                              void* d_out, int out_size, void* d_ws, size_t ws_size,
                              hipStream_t stream)
{
    const float* k_in  = (const float*)d_in[0];
    const float* v_in  = (const float*)d_in[1];
    const float* q_in  = (const float*)d_in[2];
    const unsigned char* maskp = (const unsigned char*)d_in[3];
    const float* sbias = (const float*)d_in[4];
    const float* pos_k = (const float*)d_in[5];
    const float* pos_q = (const float*)d_in[6];
    const float* Wk  = (const float*)d_in[7];   const float* bk  = (const float*)d_in[8];
    const float* Wv  = (const float*)d_in[9];   const float* bv  = (const float*)d_in[10];
    const float* Wq  = (const float*)d_in[11];  const float* bq  = (const float*)d_in[12];
    const float* Wpk = (const float*)d_in[13];  const float* bpk = (const float*)d_in[14];
    const float* Wpq = (const float*)d_in[15];  const float* bpq = (const float*)d_in[16];
    const float* Wo  = (const float*)d_in[17];  const float* bo  = (const float*)d_in[18];
    const float* Wg  = (const float*)d_in[19];  const float* bg  = (const float*)d_in[20];
    float* out = (float*)d_out;

    constexpr int Bn = 2, Ln = 2048, Dd = 1024, Hn = 16;
    const int M = Bn * Ln;
    const int rows = Bn * Hn * Ln;

    _Float16* qcat = (_Float16*)d_ws;                       // B*H*L*128
    _Float16* kcat = qcat + (size_t)rows * 128;             // B*H*L*128
    _Float16* vbuf = kcat + (size_t)rows * 128;             // B*H*L*64
    _Float16* ctx  = vbuf + (size_t)rows * 64;              // B*L*D
    float*    gatep = (float*)(ctx + (size_t)M * Dd);       // B*H*L

    dim3 blk(256);
    dim3 gproj(Dd / 128, M / 128);
    // q_cat = [q_c | q_p]
    proj_kernel<<<gproj, blk, 0, stream>>>(q_in,  Wq,  bq,  qcat, 128, 0);
    proj_kernel<<<gproj, blk, 0, stream>>>(pos_q, Wpq, bpq, qcat, 128, 64);
    // k_cat = [k_p | k_c] -> combine -> [k_c+k_p | k_c]
    proj_kernel<<<gproj, blk, 0, stream>>>(pos_k, Wpk, bpk, kcat, 128, 0);
    proj_kernel<<<gproj, blk, 0, stream>>>(k_in,  Wk,  bk,  kcat, 128, 64);
    proj_kernel<<<gproj, blk, 0, stream>>>(v_in,  Wv,  bv,  vbuf, 64, 0);

    combine_kcat<<<(rows * 64 + 255) / 256, blk, 0, stream>>>(kcat, rows);
    gate_kernel<<<(rows + 255) / 256, blk, 0, stream>>>(qcat, Wg, bg, gatep, rows);

    dim3 ga(Ln / 128, Hn, Bn);
    attn_kernel<<<ga, blk, 0, stream>>>(qcat, kcat, vbuf, gatep, sbias, maskp, ctx);

    oproj_kernel<<<gproj, blk, 0, stream>>>(ctx, Wo, bo, out);
}
